// ProbProtoSegHead_13219909337484
// MI455X (gfx1250) — matrix operand order repaired and checked by reading, not compiled, not run on hardware
//
#include <hip/hip_runtime.h>
#include <hip/hip_bf16.h>

typedef __attribute__((ext_vector_type(16))) _Float16 v16h;
typedef __attribute__((ext_vector_type(8)))  _Float16 v8h;
typedef __attribute__((ext_vector_type(8)))  float    v8f;

#define NCLS   19
#define NPROT  10
#define NSLOT  190      // 19*10
#define DIM    720
#define DIMP   736      // K padded to 23*32
#define KCH    23       // 736/32
#define CT     12       // 192/16 column tiles
#define NPIX   131072   // 4*128*256
#define SKB    240      // colsum partial blocks
#define CHUNK  547      // ceil(NPIX/SKB)
#define LSTR   201      // LDS row stride (floats), conflict-free

// ---------------------------------------------------------------- init -----
__global__ void k_init(unsigned int* wsB_u32, int nB_u32, int* nkm) {
    int i = blockIdx.x * blockDim.x + threadIdx.x;
    int stride = gridDim.x * blockDim.x;
    for (int j = i; j < nB_u32; j += stride) wsB_u32[j] = 0u;
    if (i < 192) nkm[i] = 0;
}

// ------------------------------------------------- prototype l2n + B frag --
// B-fragment layout this code uses with v_wmma_f32_16x16x32_f16 (contiguous k, same order as A):
// element e (0..15) of lane L: N = L%16, K = e + (L>=16 ? 16 : 0) within chunk.
__global__ void __launch_bounds__(256) k_protos(const float* __restrict__ protos,
                                                float* __restrict__ Pn,
                                                _Float16* __restrict__ wsB) {
    int row = blockIdx.x;           // slot 0..189 (= k*10+m)
    int t = threadIdx.x;
    int j0 = t, j1 = t + 256, j2 = t + 512;
    float p0 = protos[(size_t)row * DIM + j0];
    float p1 = protos[(size_t)row * DIM + j1];
    float p2 = (j2 < DIM) ? protos[(size_t)row * DIM + j2] : 0.f;
    __shared__ float red[256];
    red[t] = p0 * p0 + p1 * p1 + p2 * p2;
    __syncthreads();
    for (int o = 128; o > 0; o >>= 1) { if (t < o) red[t] += red[t + o]; __syncthreads(); }
    float inv = 1.f / fmaxf(sqrtf(red[0]), 1e-12f);
    int tt = row >> 4, nI = row & 15;
    float pv[3] = {p0, p1, p2};
    int   jv[3] = {j0, j1, j2};
    for (int r = 0; r < 3; r++) {
        int j = jv[r];
        if (j >= DIM) continue;
        float pn = pv[r] * inv;
        Pn[(size_t)row * DIM + j] = pn;
        int c = j >> 5, kk = j & 31;
        int lane = nI + ((kk >= 16) ? 16 : 0);
        int e = kk & 15;
        wsB[((size_t)((tt * KCH + c) * 32 + lane)) * 16 + e] = (_Float16)pn;
    }
}

// ------------------------------------------ pixel layernorm + l2n -> f16 ---
__global__ void __launch_bounds__(256) k_pixnorm(const float* __restrict__ x,
                                                 const float* __restrict__ g,
                                                 const float* __restrict__ b,
                                                 _Float16* __restrict__ wsc) {
    int w = threadIdx.x >> 5, L = threadIdx.x & 31;
    int n = blockIdx.x * 8 + w;
    const float* xp = x + (size_t)n * DIM;
    float v[22];
    float s = 0.f, sq = 0.f;
#pragma unroll
    for (int i = 0; i < 22; i++) { float tv = xp[L + 32 * i]; v[i] = tv; s += tv; sq += tv * tv; }
    float v22 = 0.f;
    if (L < 16) { v22 = xp[704 + L]; s += v22; sq += v22 * v22; }
#pragma unroll
    for (int o = 16; o > 0; o >>= 1) { s += __shfl_xor(s, o, 32); sq += __shfl_xor(sq, o, 32); }
    float mu = s * (1.f / 720.f);
    float rstd = rsqrtf(sq * (1.f / 720.f) - mu * mu + 1e-5f);
    float tv[22], tv22 = 0.f, l2 = 0.f;
#pragma unroll
    for (int i = 0; i < 22; i++) {
        int j = L + 32 * i;
        float tt = (v[i] - mu) * rstd * g[j] + b[j];
        tv[i] = tt; l2 += tt * tt;
    }
    if (L < 16) { int j = 704 + L; tv22 = (v22 - mu) * rstd * g[j] + b[j]; l2 += tv22 * tv22; }
#pragma unroll
    for (int o = 16; o > 0; o >>= 1) l2 += __shfl_xor(l2, o, 32);
    float inv = 1.f / fmaxf(sqrtf(l2), 1e-12f);
    _Float16* cp = wsc + (size_t)n * DIMP;
#pragma unroll
    for (int i = 0; i < 22; i++) cp[L + 32 * i] = (_Float16)(tv[i] * inv);
    if (L < 16) cp[704 + L] = (_Float16)(tv22 * inv);
    else        cp[720 + (L - 16)] = (_Float16)0.f;   // zero K-pad
}

// ------------------------------------------------ WMMA GEMM + fused epilog -
__global__ void __launch_bounds__(256) k_gemm(const _Float16* __restrict__ wsc,
                                              const _Float16* __restrict__ wsB,
                                              const int* __restrict__ gt,
                                              const float* __restrict__ mng, const float* __restrict__ mnb,
                                              const float* __restrict__ png, const float* __restrict__ pnb,
                                              float* __restrict__ out_seg, float* __restrict__ out_con,
                                              float* __restrict__ ws_s, unsigned char* __restrict__ ws_corr) {
    __shared__ float lds[8][8][LSTR];                       // 51456 B epilogue staging
    __shared__ __align__(32) _Float16 bsh[2][CT * 512];     // 2 x 12 KB double-buffered B
    int tid = threadIdx.x;
    int w = tid >> 5, L = tid & 31;
    int rowBase = blockIdx.x * 128 + w * 16;
    int hi = (L >= 16) ? 1 : 0;
    int M = L & 15;
    // A layout: element e of lane L: M=L%16, K = e + (L>=16?16:0)
    const _Float16* aptr = wsc + (size_t)(rowBase + M) * DIMP + (hi ? 16 : 0);

    // async-stage one k-chunk (12 fragments x 1KB) of B into LDS buffer bf
    auto stage = [&](int c, int bf) {
#pragma unroll
        for (int r = 0; r < 3; r++) {
            int p = tid + 256 * r;          // 0..767 16-byte pieces
            int tile = p >> 6;              // 0..11
            int part = p & 63;              // 16B piece within 1KB fragment
            const _Float16* gp = wsB + ((size_t)(tile * KCH + c) * 512) + part * 8;
            unsigned loff = (unsigned)(size_t)(&bsh[bf][tile * 512 + part * 8]);
            asm volatile("global_load_async_to_lds_b128 %0, %1, off"
                         :: "v"(loff), "v"(gp) : "memory");
        }
    };

    v8f acc[CT];
#pragma unroll
    for (int t = 0; t < CT; t++) { v8f z = {}; acc[t] = z; }

    stage(0, 0);
#pragma unroll 1
    for (int c = 0; c < KCH; c++) {
        int bf = c & 1;
        if (c + 1 < KCH) {
            stage(c + 1, bf ^ 1);
            asm volatile("s_wait_asynccnt 0x3" ::: "memory");  // current buffer landed
        } else {
            asm volatile("s_wait_asynccnt 0x0" ::: "memory");
        }
        __syncthreads();                                       // all waves' pieces visible
        v8h alo = *(const v8h*)(aptr + c * 32);
        v8h ahi = *(const v8h*)(aptr + c * 32 + 8);
        v16h a;
#pragma unroll
        for (int i = 0; i < 8; i++) { a[i] = alo[i]; a[8 + i] = ahi[i]; }
        const _Float16* bbase = &bsh[bf][0];
#pragma unroll
        for (int t = 0; t < CT; t++) {
            v16h bb = *(const v16h*)(bbase + t * 512 + L * 16);
            acc[t] = __builtin_amdgcn_wmma_f32_16x16x32_f16(false, a, false, bb,
                                                            (short)0, acc[t], false, false);
        }
        __syncthreads();                                       // protect bf before re-stage
    }

    const float inv190 = 1.f / 190.f, inv19 = 1.f / 19.f;
#pragma unroll
    for (int half = 0; half < 2; half++) {
        int v0 = half * 4;
        // D layout: (Mrow, Ncol) at lane = Ncol + (Mrow>=8?16:0), vgpr = Mrow%8
#pragma unroll
        for (int t = 0; t < CT; t++)
#pragma unroll
            for (int vi = 0; vi < 4; vi++) {
                int rb = vi + (hi ? 4 : 0);
                lds[w][rb][t * 16 + M] = acc[t][v0 + vi];
            }
        __syncthreads();
        int rb = L & 7, sgi = L >> 3;
        int rowM = (rb < 4) ? (v0 + rb) : (v0 + 4 + rb);
        int n1 = rowBase + rowM;
        int c0 = sgi * 48, c1 = (sgi == 3) ? 190 : (c0 + 48);
        float s = 0.f, sq = 0.f;
        for (int c = c0; c < c1; c++) { float vv = lds[w][rb][c]; s += vv; sq += vv * vv; }
        s  += __shfl_xor(s, 8, 32);  s  += __shfl_xor(s, 16, 32);
        sq += __shfl_xor(sq, 8, 32); sq += __shfl_xor(sq, 16, 32);
        float mu = s * inv190;
        float rstd = rsqrtf(sq * inv190 - mu * mu + 1e-5f);
        float* cout = out_con + (size_t)n1 * 190;
        for (int c = c0; c < c1; c++) {
            float vv = lds[w][rb][c];
            cout[c] = (vv - mu) * rstd * png[c] + pnb[c];
        }
        if (L < 8) {  // one lane per staged row: seg / argmax / sinkhorn init
            int n = n1;
            float s1 = 0.f, s2 = 0.f;
            for (int k = 0; k < NCLS; k++) {
                float mx = lds[w][rb][k * 10];
                for (int m = 1; m < 10; m++) mx = fmaxf(mx, lds[w][rb][k * 10 + m]);
                s1 += mx; s2 += mx * mx;
            }
            float mu2 = s1 * inv19;
            float rstd2 = rsqrtf(s2 * inv19 - mu2 * mu2 + 1e-5f);
            int pred = 0; float best = -3.4e38f;
            float* sout = out_seg + (size_t)n * NCLS;
            for (int k = 0; k < NCLS; k++) {
                float mx = lds[w][rb][k * 10];
                for (int m = 1; m < 10; m++) mx = fmaxf(mx, lds[w][rb][k * 10 + m]);
                float ln = (mx - mu2) * rstd2 * mng[k] + mnb[k];
                sout[k] = ln;
                if (ln > best) { best = ln; pred = k; }
            }
            int gv = gt[n];
            ws_corr[n] = (pred == gv) ? 1 : 0;
            float* sp = ws_s + (size_t)n * 10;
            for (int m = 0; m < 10; m++)
                sp[m] = __expf(lds[w][rb][gv * 10 + m] * 20.0f);   // exp(sim/0.05)
        }
        __syncthreads();
    }
}

// ----------------------------------------------- Sinkhorn: colsum (2-stage) -
__global__ void __launch_bounds__(192) k_colsum_part(const int* __restrict__ gt,
                                                     const float* __restrict__ ws_s,
                                                     float* __restrict__ part) {
    int b = blockIdx.x, j = threadIdx.x;
    int n0 = b * CHUNK, n1 = n0 + CHUNK; if (n1 > NPIX) n1 = NPIX;
    float acc = 0.f;
    if (j < NSLOT) {
        int k = j / 10, m = j - 10 * k;
        for (int n = n0; n < n1; n++)
            if (gt[n] == k) acc += ws_s[(size_t)n * 10 + m];
    }
    part[b * 192 + j] = acc;
}

__global__ void __launch_bounds__(192) k_colsum_fin(const float* __restrict__ part,
                                                    float* __restrict__ colsum) {
    int j = threadIdx.x;
    float s = 0.f;
    for (int b = 0; b < SKB; b++) s += part[b * 192 + j];
    colsum[j] = s;
}

// --------------------------------------- Sinkhorn row-normalize / finalize --
__global__ void __launch_bounds__(256) k_sk(const int* __restrict__ gt,
                                            const float* __restrict__ colsum,
                                            float* __restrict__ ws_s,
                                            const unsigned char* __restrict__ ws_corr,
                                            float* __restrict__ proto_target,
                                            int* __restrict__ ws_slot,
                                            int* __restrict__ nkm, int last) {
    int n = blockIdx.x * blockDim.x + threadIdx.x;
    if (n >= NPIX) return;
    int g = gt[n];
    const float* cs = colsum + g * 10;
    float* sp = ws_s + (size_t)n * 10;
    float r[10]; float rs = 0.f;
#pragma unroll
    for (int m = 0; m < 10; m++) {
        float u = sp[m];
        float c = cs[m];
        u = (c > 0.f) ? u / c : 0.f;
        r[m] = u; rs += u;
    }
    float rinv = (rs > 0.f) ? 1.f / rs : 0.f;
    if (!last) {
#pragma unroll
        for (int m = 0; m < 10; m++) sp[m] = r[m] * rinv;
    } else {
        int idx = 0; float best = r[0];
#pragma unroll
        for (int m = 1; m < 10; m++) if (r[m] > best) { best = r[m]; idx = m; }
        proto_target[n] = (float)(idx + 10 * g);
        int slot = g * 10 + idx;
        int corr = ws_corr[n];
        ws_slot[n] = corr ? slot : -1;
        if (corr) atomicAdd(&nkm[slot], 1);
    }
}

// ------------------------------------------ prototype EMA update + l2n ------
__global__ void __launch_bounds__(256) k_protoupd(const int* __restrict__ ws_slot,
                                                  const _Float16* __restrict__ wsc,
                                                  const float* __restrict__ Pn,
                                                  const int* __restrict__ nkm,
                                                  float* __restrict__ out_pr) {
    int s = blockIdx.x;                       // slot 0..189
    int w = threadIdx.x >> 5, L = threadIdx.x & 31;
    int t = threadIdx.x;
    float acc[23];
#pragma unroll
    for (int i = 0; i < 23; i++) acc[i] = 0.f;
    int n0 = w * (NPIX / 8), n1 = n0 + NPIX / 8;
    for (int n = n0; n < n1; n += 4) {
        int4 sl = *(const int4*)(ws_slot + n);
        int sls[4] = {sl.x, sl.y, sl.z, sl.w};
#pragma unroll
        for (int q = 0; q < 4; q++) {
            if (sls[q] == s) {
                const _Float16* cp = wsc + (size_t)(n + q) * DIMP;
#pragma unroll
                for (int i = 0; i < 22; i++) acc[i] += (float)cp[L + 32 * i];
                if (L < 16) acc[22] += (float)cp[704 + L];
            }
        }
    }
    __shared__ float fpar[8][DIM];
    __shared__ float ftot[DIM];
    __shared__ float red[256];
#pragma unroll
    for (int i = 0; i < 22; i++) fpar[w][L + 32 * i] = acc[i];
    if (L < 16) fpar[w][704 + L] = acc[22];
    __syncthreads();
    float ss = 0.f;
    for (int rep = 0; rep < 3; rep++) {
        int j = t + rep * 256;
        if (j < DIM) {
            float f = 0.f;
#pragma unroll
            for (int wi = 0; wi < 8; wi++) f += fpar[wi][j];
            ftot[j] = f; ss += f * f;
        }
    }
    red[t] = ss; __syncthreads();
    for (int o = 128; o > 0; o >>= 1) { if (t < o) red[t] += red[t + o]; __syncthreads(); }
    float finv = 1.f / fmaxf(sqrtf(red[0]), 1e-12f);
    int valid = nkm[s] > 0;
    float bb[3]; float bs = 0.f;
    for (int rep = 0; rep < 3; rep++) {
        int j = t + rep * 256;
        float bv = 0.f;
        if (j < DIM) {
            float pv = Pn[(size_t)s * DIM + j];
            float fn = ftot[j] * finv;
            bv = valid ? (0.999f * pv + 0.001f * fn) : pv;
        }
        bb[rep] = bv; bs += bv * bv;
    }
    __syncthreads();
    red[t] = bs; __syncthreads();
    for (int o = 128; o > 0; o >>= 1) { if (t < o) red[t] += red[t + o]; __syncthreads(); }
    float binv = 1.f / fmaxf(sqrtf(red[0]), 1e-12f);
    for (int rep = 0; rep < 3; rep++) {
        int j = t + rep * 256;
        if (j < DIM) out_pr[(size_t)s * DIM + j] = bb[rep] * binv;
    }
}

// ---------------------------------------------------------------- launch ---
extern "C" void kernel_launch(void* const* d_in, const int* in_sizes, int n_in,
                              void* d_out, int out_size, void* d_ws, size_t ws_size,
                              hipStream_t stream) {
    const float* x      = (const float*)d_in[0];
    const float* protos = (const float*)d_in[1];
    const float* fn_g   = (const float*)d_in[2];
    const float* fn_b   = (const float*)d_in[3];
    const float* mn_g   = (const float*)d_in[4];
    const float* mn_b   = (const float*)d_in[5];
    const float* pn_g   = (const float*)d_in[6];
    const float* pn_b   = (const float*)d_in[7];
    const int*   gt     = (const int*)d_in[8];

    float* out_seg = (float*)d_out;                       // [N,19]
    float* out_con = out_seg + (size_t)NPIX * NCLS;       // [N,190]
    float* out_pt  = out_con + (size_t)NPIX * NSLOT;      // [N]
    float* out_pr  = out_pt + NPIX;                       // [19,10,720]

    char* wp = (char*)d_ws;
    size_t off = 0;
    _Float16* ws_c   = (_Float16*)(wp + off); off += (size_t)NPIX * DIMP * 2;
    float*    ws_s   = (float*)(wp + off);    off += (size_t)NPIX * 10 * 4;
    int*      ws_slot= (int*)(wp + off);      off += (size_t)NPIX * 4;
    unsigned char* ws_corr = (unsigned char*)(wp + off); off += NPIX;
    float*    ws_Pn  = (float*)(wp + off);    off += (size_t)NSLOT * DIM * 4;
    _Float16* ws_B   = (_Float16*)(wp + off); off += (size_t)CT * KCH * 32 * 16 * 2;
    float*    ws_part= (float*)(wp + off);    off += (size_t)SKB * 192 * 4;
    float*    ws_cs  = (float*)(wp + off);    off += 192 * 4;
    int*      ws_nkm = (int*)(wp + off);      off += 192 * 4;

    int nB_u32 = (CT * KCH * 32 * 16 * 2) / 4;
    k_init<<<288, 256, 0, stream>>>((unsigned int*)ws_B, nB_u32, ws_nkm);
    k_protos<<<NSLOT, 256, 0, stream>>>(protos, ws_Pn, ws_B);
    k_pixnorm<<<NPIX / 8, 256, 0, stream>>>(x, fn_g, fn_b, ws_c);
    k_gemm<<<NPIX / 128, 256, 0, stream>>>(ws_c, ws_B, gt, mn_g, mn_b, pn_g, pn_b,
                                           out_seg, out_con, ws_s, ws_corr);
    for (int it = 0; it < 3; it++) {
        k_colsum_part<<<SKB, 192, 0, stream>>>(gt, ws_s, ws_part);
        k_colsum_fin<<<1, 192, 0, stream>>>(ws_part, ws_cs);
        k_sk<<<NPIX / 256, 256, 0, stream>>>(gt, ws_cs, ws_s, ws_corr,
                                             out_pt, ws_slot, ws_nkm, (it == 2) ? 1 : 0);
    }
    k_protoupd<<<NSLOT, 256, 0, stream>>>(ws_slot, ws_c, ws_Pn, ws_nkm, out_pr);
}
